// SparseExperienceEncoder_87677462381242
// MI455X (gfx1250) — matrix operand order repaired and checked by reading, not compiled, not run on hardware
//
#include <hip/hip_runtime.h>

#define B_ROWS 8192
#define EXP_D  1024
#define DICT_K 4096
#define N_IT   20
#define K_SP   409
#define STEPSZ 0.1f
#define THRESH 0.01f   /* REG * STEP */

typedef __attribute__((ext_vector_type(16))) __bf16        v16bf;
typedef __attribute__((ext_vector_type(8)))  float         v8f;
typedef __attribute__((ext_vector_type(4)))  unsigned int  v4u;
typedef __attribute__((ext_vector_type(4)))  int           v4i;

union FragBF { v16bf h; v4u q[2]; };

__device__ __forceinline__ unsigned short f2bf(float f) {
  unsigned int u = __float_as_uint(f);
  u += 0x7FFFu + ((u >> 16) & 1u);   // round-to-nearest-even
  return (unsigned short)(u >> 16);
}

// ---------------------------------------------------------------------------
// Async global->LDS staging (CDNA5 GLOBAL_LOAD_ASYNC_TO_LDS_B128, ASYNCcnt).
// Builtin signature (from hipcc diagnostic): arg0 = v4i addrspace(1)*,
// arg1 = v4i addrspace(3)*, then imm offset, imm cpol.
// ---------------------------------------------------------------------------
#if defined(__HIP_DEVICE_COMPILE__) && \
    __has_builtin(__builtin_amdgcn_global_load_async_to_lds_b128) && \
    __has_builtin(__builtin_amdgcn_s_wait_asynccnt)
#define USE_ASYNC_LDS 1
#else
#define USE_ASYNC_LDS 0
#endif

#if USE_ASYNC_LDS
typedef __attribute__((address_space(1))) v4i* gas_v4i_p;
typedef __attribute__((address_space(3))) v4i* las_v4i_p;
#endif

__device__ __forceinline__ void stage16(const unsigned short* gp, unsigned short* lp) {
#if USE_ASYNC_LDS
  __builtin_amdgcn_global_load_async_to_lds_b128(
      (gas_v4i_p)(unsigned short*)gp, (las_v4i_p)lp, 0, 0);
#else
  *(v4u*)lp = *(const v4u*)gp;
#endif
}
__device__ __forceinline__ void stage_wait() {
#if USE_ASYNC_LDS
  __builtin_amdgcn_s_wait_asynccnt(0);
#endif
}

// ---------------------------------------------------------------------------
// Tiled bf16 WMMA GEMM, 128x128 block tile, 8 wave32 waves, K-step 32.
// A (iteration-varying) goes through LDS; B comes pre-swizzled from global in
// WMMA-fragment-ready layout: fragment fi=(n0/16)*(KRED/32)+(kk/32) stored as
// 512 contiguous halves addressed by lane*16+h  (h: K=(lane>=16?16:0)+h,
// N = n0 + (lane&15)), so a B fragment is two contiguous global b128 loads.
// MODE 0: out_bf16 = bf16( Cf - A@B )            (residual = X - alpha@D)
// MODE 1: a = Cf + 0.1*(A@B); soft-threshold; write f32 + bf16 alpha
// ---------------------------------------------------------------------------
template<int MODE, int KRED, int NDIM>
__launch_bounds__(256)
__global__ void gemm_bf16_k(const unsigned short* __restrict__ A,
                            const unsigned short* __restrict__ Bswz,
                            const float*          __restrict__ Cf,
                            float*                __restrict__ outF,
                            unsigned short*       __restrict__ outB)
{
  __shared__ __align__(16) unsigned short sA[128 * 40]; // [row][k], stride 40 halves

  const int t    = threadIdx.x;
  const int lane = t & 31;
  const int wid  = t >> 5;
  const int l15  = lane & 15;
  const int lh   = lane >> 4;
  const int mOff = (wid & 3) << 5;          // 4 waves along M: 0,32,64,96
  const int nOff = (wid >> 2) << 6;         // 2 waves along N: 0,64
  const int mBlock = blockIdx.y << 7;
  const int nBlock = blockIdx.x << 7;

  // A staging: 128 rows x 32 halves = 512 b128 chunks; thread does 2 chunks.
  const int arow = t >> 2;                  // 0..63
  const int acol = (t & 3) << 3;            // 0,8,16,24

  // Per-wave B fragment bases (constant part), lane-addressed.
  const unsigned short* bbase[4];
#pragma unroll
  for (int ni = 0; ni < 4; ++ni) {
    size_t fcol = (size_t)((nBlock + nOff + (ni << 4)) >> 4);
    bbase[ni] = Bswz + ((fcol * (KRED >> 5)) << 9) + (lane << 4);
  }

  v8f acc[2][4] = {};

  for (int kk = 0; kk < KRED; kk += 32) {
    __syncthreads();                        // protect previous tile's readers
    {
      const unsigned short* gA0 = A + (size_t)(mBlock + arow) * KRED + kk + acol;
      const unsigned short* gA1 = gA0 + (size_t)64 * KRED;
      stage16(gA0, &sA[arow * 40 + acol]);
      stage16(gA1, &sA[(arow + 64) * 40 + acol]);
      if (kk + 32 < KRED) {                 // uniform branch
        __builtin_prefetch(gA0 + 32, 0, 1); // global_prefetch_b8
        __builtin_prefetch(gA1 + 32, 0, 1);
      }
    }
    stage_wait();
    __syncthreads();

    FragBF fa[2], fb[4];
#pragma unroll
    for (int mi = 0; mi < 2; ++mi) {
      int ar = mOff + (mi << 4) + l15;      // M = lane&15
      int ak = lh << 4;                     // halves 0-15: K=(lh?16:0)+h
      fa[mi].q[0] = *(const v4u*)(&sA[ar * 40 + ak]);
      fa[mi].q[1] = *(const v4u*)(&sA[ar * 40 + ak + 8]);
    }
#pragma unroll
    for (int ni = 0; ni < 4; ++ni) {
      const unsigned short* bp = bbase[ni] + ((size_t)(kk >> 5) << 9);
      fb[ni].q[0] = *(const v4u*)bp;
      fb[ni].q[1] = *(const v4u*)(bp + 8);
      if (kk + 32 < KRED) __builtin_prefetch(bp + 512, 0, 1);
    }
#pragma unroll
    for (int mi = 0; mi < 2; ++mi)
#pragma unroll
      for (int ni = 0; ni < 4; ++ni)
        acc[mi][ni] = __builtin_amdgcn_wmma_f32_16x16x32_bf16(
            false, fa[mi].h, false, fb[ni].h, (short)0, acc[mi][ni], false, false);
  }

  // ---- fused epilogue (C/D layout: VGPR v -> M = v + 8*lh, lane&15 -> N) ----
#pragma unroll
  for (int mi = 0; mi < 2; ++mi) {
#pragma unroll
    for (int ni = 0; ni < 4; ++ni) {
#pragma unroll
      for (int v = 0; v < 8; ++v) {
        int row = mBlock + mOff + (mi << 4) + (lh << 3) + v;
        int col = nBlock + nOff + (ni << 4) + l15;
        size_t idx = (size_t)row * NDIM + col;
        float a = acc[mi][ni][v];
        if (MODE == 0) {
          outB[idx] = f2bf(Cf[idx] - a);               // residual, bf16
        } else {
          float al = Cf[idx] + STEPSZ * a;
          float s  = fabsf(al) - THRESH;
          float o  = (s > 0.f) ? copysignf(s, al) : 0.f;
          outF[idx] = o;                               // f32 master alpha
          outB[idx] = f2bf(o);                         // bf16 shadow for next GEMM
        }
      }
    }
  }
}

// ---------------------------------------------------------------------------
// Setup: convert dictionary once into the two fragment-ready B layouts.
//   dswz : B for residual GEMM  = D   as [KRED=4096][N=1024]
//   dtswz: B for update   GEMM  = D^T as [KRED=1024][N=4096]
// ---------------------------------------------------------------------------
__global__ void cvt_d_k(const float* __restrict__ d,
                        unsigned short* __restrict__ dswz,
                        unsigned short* __restrict__ dtswz)
{
  int i = blockIdx.x * 256 + threadIdx.x;
  if (i >= DICT_K * EXP_D) return;
  int k = i >> 10;          // atom   0..4095
  int j = i & 1023;         // dim    0..1023
  unsigned short v = f2bf(d[i]);
  {   // residual GEMM: element (K=k, N=j), KRED=4096
    int lh = (k >> 4) & 1, h = k & 15, l = (lh << 4) | (j & 15);
    size_t fi = (size_t)(j >> 4) * (DICT_K >> 5) + (k >> 5);
    dswz[(fi << 9) + (l << 4) + h] = v;
  }
  {   // update GEMM: element (K=j, N=k), KRED=1024
    int lh = (j >> 4) & 1, h = j & 15, l = (lh << 4) | (k & 15);
    size_t fi = (size_t)(k >> 4) * (EXP_D >> 5) + (j >> 5);
    dtswz[(fi << 9) + (l << 4) + h] = v;
  }
}

__global__ void zero_k(float* __restrict__ aF, unsigned short* __restrict__ aB, size_t n) {
  size_t i = (size_t)blockIdx.x * 256 + threadIdx.x;
  if (i < n) { aF[i] = 0.f; aB[i] = 0; }
}

// ---------------------------------------------------------------------------
// Per-row top-409 hard sparsification, in place. One block per row.
// ---------------------------------------------------------------------------
__launch_bounds__(256)
__global__ void topk_k(float* __restrict__ alpha) {
  __shared__ float    sv[DICT_K];
  __shared__ unsigned scnt;
  __shared__ unsigned sscan[256];
  const int t = threadIdx.x;
  float* rp = alpha + (size_t)blockIdx.x * DICT_K;
  for (int i = t; i < DICT_K; i += 256) sv[i] = rp[i];
  __syncthreads();

  unsigned lo = 0u, hi = 0x7f7fffffu;
  while (lo < hi) {                      // uniform across block: ~31 iters
    unsigned mid = lo + ((hi - lo + 1) >> 1);
    if (t == 0) scnt = 0;
    __syncthreads();
    unsigned c = 0;
    for (int i = t * 16; i < t * 16 + 16; ++i)
      c += ((__float_as_uint(sv[i]) & 0x7fffffffu) >= mid);
    atomicAdd(&scnt, c);
    __syncthreads();
    unsigned tot = scnt;
    if (tot >= K_SP) lo = mid; else hi = mid - 1;
    __syncthreads();
  }
  const unsigned thr = lo;

  if (t == 0) scnt = 0;
  __syncthreads();
  unsigned cg = 0, ce = 0;
  for (int i = t * 16; i < t * 16 + 16; ++i) {
    unsigned ab = __float_as_uint(sv[i]) & 0x7fffffffu;
    cg += (ab > thr);
    ce += (ab == thr);
  }
  atomicAdd(&scnt, cg);
  sscan[t] = ce;
  __syncthreads();
  const unsigned nG    = scnt;
  const unsigned quota = (K_SP > nG) ? (K_SP - nG) : 0u;

  for (int off = 1; off < 256; off <<= 1) {   // inclusive scan of equal-counts
    unsigned v = sscan[t];
    unsigned a = (t >= off) ? sscan[t - off] : 0u;
    __syncthreads();
    sscan[t] = v + a;
    __syncthreads();
  }
  unsigned rank = sscan[t] - ce;              // exclusive prefix

  for (int i = t * 16; i < t * 16 + 16; ++i) {
    float v = sv[i];
    unsigned ab = __float_as_uint(v) & 0x7fffffffu;
    bool keep = false;
    if (ab > thr) keep = true;
    else if (ab == thr) { if (rank < quota) keep = true; rank++; }
    rp[i] = keep ? v : 0.f;
  }
}

// ---------------------------------------------------------------------------
extern "C" void kernel_launch(void* const* d_in, const int* in_sizes, int n_in,
                              void* d_out, int out_size, void* d_ws, size_t ws_size,
                              hipStream_t stream) {
  const float* X = (const float*)d_in[0];          // [8192,1024]
  const float* D = (const float*)d_in[1];          // [4096,1024]
  float* alphaF  = (float*)d_out;                  // [8192,4096] master alpha

  char* ws = (char*)d_ws;
  unsigned short* Dswz  = (unsigned short*)ws;  ws += (size_t)DICT_K * EXP_D  * 2;  //  8 MB
  unsigned short* Dtswz = (unsigned short*)ws;  ws += (size_t)EXP_D  * DICT_K * 2;  //  8 MB
  unsigned short* Ab    = (unsigned short*)ws;  ws += (size_t)B_ROWS * DICT_K * 2;  // 64 MB
  unsigned short* Rb    = (unsigned short*)ws;                                      // 16 MB

  cvt_d_k<<<(DICT_K * EXP_D + 255) / 256, 256, 0, stream>>>(D, Dswz, Dtswz);
  const size_t nA = (size_t)B_ROWS * DICT_K;
  zero_k<<<(unsigned)((nA + 255) / 256), 256, 0, stream>>>(alphaF, Ab, nA);

  dim3 blk(256, 1, 1);
  dim3 gR(EXP_D  / 128, B_ROWS / 128, 1);   // residual GEMM: ( 8, 64)
  dim3 gU(DICT_K / 128, B_ROWS / 128, 1);   // update   GEMM: (32, 64)
  for (int it = 0; it < N_IT; ++it) {
    // residual = X - alpha @ D   (A = alpha_bf16 [8192x4096], B = D swizzled)
    gemm_bf16_k<0, DICT_K, EXP_D><<<gR, blk, 0, stream>>>(Ab, Dswz, X, nullptr, Rb);
    // alpha = soft(alpha + 0.1 * residual @ D^T)
    gemm_bf16_k<1, EXP_D, DICT_K><<<gU, blk, 0, stream>>>(Rb, Dtswz, alphaF, alphaF, Ab);
  }

  topk_k<<<B_ROWS, 256, 0, stream>>>(alphaF);
}
